// SeqDeformableEncoderLayer_15444702396600
// MI455X (gfx1250) — matrix operand order repaired and checked by reading, not compiled, not run on hardware
//
#include <hip/hip_runtime.h>
#include <math.h>

// ---------------- problem constants ----------------
#define L_     3
#define H_     180
#define W_     180
#define HW_    32400
#define C_     128
#define HEADS_ 8
#define P_     4
#define DFF_   1024
#define NTOK   97200   // L_*HW_

typedef _Float16 half_t;
typedef __attribute__((ext_vector_type(16))) half_t v16h;
typedef __attribute__((ext_vector_type(8)))  float  v8f;
typedef __attribute__((ext_vector_type(4)))  int    v4i;

union AView { v16h h; v4i i[2]; };

// ---------------- packed-weight segment offsets (in halfs) ----------------
#define PK_CONV 0        // 9 taps * (KT=8,NT=8) * 512 = 294912
#define PK_VAL  294912   // (KT=4,NT=8)*512  = 16384
#define PK_COMB 311296   // (KT=4,NT=6)*512  = 12288   (W_off | W_attn)
#define PK_OUT  323584   // 16384
#define PK_W1   339968   // (KT=4,NT=64)*512 = 131072
#define PK_W2   471040   // (KT=32,NT=8)*512 = 131072
#define PK_TOT  602112

// ---------------- workspace byte offsets (aliased where safe) -------------
#define WS_PACK   ((size_t)0)          //  1.2 MB packed f16 weights
#define WS_SRC16  ((size_t)2097152)    // 24.9 MB f16 src_all
#define WS_Q16    ((size_t)26980352)   // 24.9 MB f16 queries
#define WS_VALUE  ((size_t)51863552)   // 49.8 MB f32 value [L][8][HW][16]
#define WS_OFFATT ((size_t)101629952)  // 37.3 MB f32 [L*HW][96]
#define WS_OUT16  ((size_t)138954752)  //  8.3 MB f16 attn output [HW][128]
#define WS_SRC2   ((size_t)147249152)  // 16.6 MB f32
#define WS_SRCF   ((size_t)163837952)  // 16.6 MB f32 post-LN1
#define WS_SRCLN  ((size_t)180426752)  //  8.3 MB f16 post-LN1
#define WS_HID16  WS_SRC16             // 66.4 MB f16 FFN hidden (aliases src16/q16/value, all dead by then)
#define WS_FFN    WS_OFFATT            // 16.6 MB f32 FFN out (aliases offatt, dead by then)

// ---------------- WMMA helpers ----------------
__device__ __forceinline__ v16h load_a16(const half_t* A, int row, int ldk, int kt, int hf) {
  // 16-bit A layout: lane m = L%16, half = L>>4; elems 0..7 -> K = hf*8+e ; 8..15 -> K = 16+hf*8+(e-8)
  const half_t* ap = A + (size_t)row * ldk + kt * 32 + hf * 8;
  AView av;
  av.i[0] = *(const v4i*)(ap);
  av.i[1] = *(const v4i*)(ap + 16);
  return av.h;
}
__device__ __forceinline__ v16h load_bpk(const half_t* P, int kt, int nt, int KT, int lane) {
  return ((const v16h*)P)[(size_t)(nt * KT + kt) * 32 + lane];
}
__device__ __forceinline__ v8f wmma_f16(v16h a, v16h b, v8f c) {
  return __builtin_amdgcn_wmma_f32_16x16x32_f16(false, a, false, b, (short)0, c, false, false);
}

// ---------------- weight packing ----------------
__device__ __forceinline__ void pk_idx(int r, int KT, int& k, int& n) {
  int e = r & 15, lane = (r >> 4) & 31, tile = r >> 9;
  int kt = tile % KT, nt = tile / KT;
  k = kt * 32 + ((e >> 3) << 4) + ((lane >> 4) << 3) + (e & 7);   // B layout: elems 0..7 -> K = hf*8+e ; 8..15 -> K = 16+hf*8+(e-8)
  n = nt * 16 + (lane & 15);
}

__global__ void k_prep(const float* conv_w, const float* W_val, const float* W_off,
                       const float* W_attn, const float* W_out, const float* W1,
                       const float* W2, half_t* pack) {
  for (int u = blockIdx.x * blockDim.x + threadIdx.x; u < PK_TOT; u += gridDim.x * blockDim.x) {
    float val;
    if (u < PK_VAL) {                       // conv taps: B_t[k][n] = conv_w[n][k][ky][kx]
      int t = u / 32768, r = u % 32768, k, n;
      pk_idx(r, 8, k, n);
      val = conv_w[(size_t)(n * 256 + k) * 9 + t];
    } else if (u < PK_COMB) {
      int k, n; pk_idx(u - PK_VAL, 4, k, n);
      val = W_val[k * 128 + n];
    } else if (u < PK_OUT) {
      int k, n; pk_idx(u - PK_COMB, 4, k, n);
      val = (n < 64) ? W_off[k * 64 + n] : W_attn[k * 32 + (n - 64)];
    } else if (u < PK_W1) {
      int k, n; pk_idx(u - PK_OUT, 4, k, n);
      val = W_out[k * 128 + n];
    } else if (u < PK_W2) {
      int k, n; pk_idx(u - PK_W1, 4, k, n);
      val = W1[k * 1024 + n];
    } else {
      int k, n; pk_idx(u - PK_W2, 32, k, n);
      val = W2[k * 128 + n];
    }
    pack[u] = (half_t)val;
  }
}

// ---------------- f32 -> f16 convert ----------------
__global__ void k_cvt(const float* src, half_t* dst, int n) {
  for (int i = blockIdx.x * blockDim.x + threadIdx.x; i < n; i += gridDim.x * blockDim.x)
    dst[i] = (half_t)src[i];
}

// ---------------- query generator: 3x3 conv as 9 GEMM taps --------------
// grid (2025, 1, 3), block 64 (2 waves x 64 cols)
__global__ void k_conv(const half_t* src16, const half_t* pack, const float* conv_b,
                       half_t* q16) {
  int lane = threadIdx.x & 31, wid = threadIdx.x >> 5;
  int m = lane & 15, hf = lane >> 4;
  int p0 = blockIdx.x * 16, l = blockIdx.z;
  int nb = wid * 64;
  int p = p0 + m, y = p / W_, x = p - y * W_;
  const half_t* f0 = src16;
  const half_t* fl = src16 + (size_t)l * HW_ * C_;
  v8f acc[4] = {};
#pragma unroll
  for (int ky = 0; ky < 3; ky++) {
#pragma unroll
    for (int kx = 0; kx < 3; kx++) {
      int yy = y + ky - 1, xx = x + kx - 1;
      bool valid = ((unsigned)yy < (unsigned)H_) && ((unsigned)xx < (unsigned)W_);
      int pn = yy * W_ + xx;
      const half_t* pt = pack + (size_t)(ky * 3 + kx) * 32768;
#pragma unroll
      for (int kt = 0; kt < 8; kt++) {
        v16h a = {};
        if (valid) {
          const half_t* rp = (kt < 4 ? f0 + (size_t)pn * C_ + kt * 32
                                     : fl + (size_t)pn * C_ + (kt - 4) * 32) + hf * 8;
          AView av;
          av.i[0] = *(const v4i*)(rp);
          av.i[1] = *(const v4i*)(rp + 16);
          a = av.h;
        }
#pragma unroll
        for (int j = 0; j < 4; j++)
          acc[j] = wmma_f16(a, load_bpk(pt, kt, (nb >> 4) + j, 8, lane), acc[j]);
      }
    }
  }
  half_t* qb = q16 + (size_t)l * HW_ * C_;
#pragma unroll
  for (int j = 0; j < 4; j++) {
    int n = nb + j * 16 + (lane & 15);
    float bias = conv_b[n];
#pragma unroll
    for (int r = 0; r < 8; r++) {
      int prow = p0 + hf * 8 + r;
      qb[(size_t)prow * C_ + n] = (half_t)(acc[j][r] + bias);
    }
  }
}

// ---------------- value projection: [97200,128]x[128,128] ----------------
// grid 6075, block 64
__global__ void k_val(const half_t* src16, const half_t* pack, const float* b_val,
                      float* value) {
  int lane = threadIdx.x & 31, wid = threadIdx.x >> 5;
  int m = lane & 15, hf = lane >> 4;
  int r0 = blockIdx.x * 16, nb = wid * 64;
  const half_t* P = pack + PK_VAL;
  v8f acc[4] = {};
#pragma unroll
  for (int kt = 0; kt < 4; kt++) {
    v16h a = load_a16(src16, r0 + m, C_, kt, hf);
#pragma unroll
    for (int j = 0; j < 4; j++)
      acc[j] = wmma_f16(a, load_bpk(P, kt, (nb >> 4) + j, 4, lane), acc[j]);
  }
#pragma unroll
  for (int j = 0; j < 4; j++) {
    int n = nb + j * 16 + (lane & 15);
    float bias = b_val[n];
    int h = n >> 4, d = n & 15;
#pragma unroll
    for (int r = 0; r < 8; r++) {
      int tok = r0 + hf * 8 + r;
      int l = tok / HW_, pq = tok - l * HW_;
      value[(((size_t)(l * HEADS_ + h)) * HW_ + pq) * 16 + d] = acc[j][r] + bias;
    }
  }
}

// ------------- fused offsets+attn logits: [97200,128]x[128,96] -----------
// grid 6075, block 64 (2 waves x 48 cols)
__global__ void k_offattn(const half_t* q16, const half_t* pack, const float* b_off,
                          const float* b_attn, float* offatt) {
  int lane = threadIdx.x & 31, wid = threadIdx.x >> 5;
  int m = lane & 15, hf = lane >> 4;
  int r0 = blockIdx.x * 16, nb = wid * 48;
  const half_t* P = pack + PK_COMB;
  v8f acc[3] = {};
#pragma unroll
  for (int kt = 0; kt < 4; kt++) {
    v16h a = load_a16(q16, r0 + m, C_, kt, hf);
#pragma unroll
    for (int j = 0; j < 3; j++)
      acc[j] = wmma_f16(a, load_bpk(P, kt, (nb >> 4) + j, 4, lane), acc[j]);
  }
#pragma unroll
  for (int j = 0; j < 3; j++) {
    int n = nb + j * 16 + (lane & 15);
    float bias = (n < 64) ? b_off[n] : b_attn[n - 64];
#pragma unroll
    for (int r = 0; r < 8; r++) {
      int tok = r0 + hf * 8 + r;
      offatt[(size_t)tok * 96 + n] = acc[j][r] + bias;
    }
  }
}

// ------------- softmax + bilinear sampling + weighted sum ----------------
// thread per (query, head); grid 1013 x 256
__global__ void k_sample(const float* offatt, const float* value, const float* refp,
                         half_t* out16) {
  int idx = blockIdx.x * blockDim.x + threadIdx.x;
  if (idx >= HW_ * HEADS_) return;
  int q = idx >> 3, h = idx & 7;

  float lg[12], wt[12];
#pragma unroll
  for (int l = 0; l < 3; l++)
#pragma unroll
    for (int p = 0; p < 4; p++)
      lg[l * 4 + p] = offatt[((size_t)(l * HW_ + q)) * 96 + 64 + h * 4 + p];
  float mx = lg[0];
#pragma unroll
  for (int i = 1; i < 12; i++) mx = fmaxf(mx, lg[i]);
  float sum = 0.f;
#pragma unroll
  for (int i = 0; i < 12; i++) { wt[i] = __expf(lg[i] - mx); sum += wt[i]; }
  float inv = 1.f / sum;

  float acc[16];
#pragma unroll
  for (int d = 0; d < 16; d++) acc[d] = 0.f;

  for (int l = 0; l < 3; l++) {
    float rx = refp[(q * 3 + l) * 2 + 0];
    float ry = refp[(q * 3 + l) * 2 + 1];
    const float* vb = value + ((size_t)(l * HEADS_ + h)) * HW_ * 16;
    const float* oa = offatt + ((size_t)(l * HW_ + q)) * 96 + h * 8;
#pragma unroll
    for (int p = 0; p < 4; p++) {
      float px = (rx + oa[p * 2 + 0] * (1.0f / 180.0f)) * (float)W_ - 0.5f;
      float py = (ry + oa[p * 2 + 1] * (1.0f / 180.0f)) * (float)H_ - 0.5f;
      float fx0 = floorf(px), fy0 = floorf(py);
      int x0 = (int)fx0, y0 = (int)fy0;
      float fx = px - fx0, fy = py - fy0;
      float aw = wt[l * 4 + p] * inv;
      float cw[4] = { (1.f - fx) * (1.f - fy) * aw, fx * (1.f - fy) * aw,
                      (1.f - fx) * fy * aw,         fx * fy * aw };
      int cx[4] = { x0, x0 + 1, x0, x0 + 1 };
      int cy[4] = { y0, y0, y0 + 1, y0 + 1 };
#pragma unroll
      for (int c = 0; c < 4; c++) {
        if ((unsigned)cx[c] < (unsigned)W_ && (unsigned)cy[c] < (unsigned)H_) {
          const float4* vr = (const float4*)(vb + ((size_t)(cy[c] * W_ + cx[c])) * 16);
          float k = cw[c];
#pragma unroll
          for (int jj = 0; jj < 4; jj++) {
            float4 vv = vr[jj];
            acc[jj * 4 + 0] += k * vv.x; acc[jj * 4 + 1] += k * vv.y;
            acc[jj * 4 + 2] += k * vv.z; acc[jj * 4 + 3] += k * vv.w;
          }
        }
      }
    }
  }
#pragma unroll
  for (int d = 0; d < 16; d++)
    out16[(size_t)q * C_ + h * 16 + d] = (half_t)acc[d];
}

// ---------------- output projection: [32400,128]x[128,128] ---------------
// grid 2025, block 64
__global__ void k_outproj(const half_t* out16, const half_t* pack, const float* b_out,
                          float* src2) {
  int lane = threadIdx.x & 31, wid = threadIdx.x >> 5;
  int m = lane & 15, hf = lane >> 4;
  int r0 = blockIdx.x * 16, nb = wid * 64;
  const half_t* P = pack + PK_OUT;
  v8f acc[4] = {};
#pragma unroll
  for (int kt = 0; kt < 4; kt++) {
    v16h a = load_a16(out16, r0 + m, C_, kt, hf);
#pragma unroll
    for (int j = 0; j < 4; j++)
      acc[j] = wmma_f16(a, load_bpk(P, kt, (nb >> 4) + j, 4, lane), acc[j]);
  }
#pragma unroll
  for (int j = 0; j < 4; j++) {
    int n = nb + j * 16 + (lane & 15);
    float bias = b_out[n];
#pragma unroll
    for (int r = 0; r < 8; r++)
      src2[(size_t)(r0 + hf * 8 + r) * C_ + n] = acc[j][r] + bias;
  }
}

// ---------------- residual + layernorm (one wave per row) ----------------
// grid 4050 x 256 (8 rows / block); yh==nullptr for the final LN
__global__ void k_ln(const float* xa, const float* xb, const float* g, const float* bt,
                     float* yf, half_t* yh) {
  int row = blockIdx.x * 8 + (threadIdx.x >> 5);
  int lane = threadIdx.x & 31;
  const float4 a4 = ((const float4*)(xa + (size_t)row * C_))[lane];
  const float4 b4 = ((const float4*)(xb + (size_t)row * C_))[lane];
  float x0 = a4.x + b4.x, x1 = a4.y + b4.y, x2 = a4.z + b4.z, x3 = a4.w + b4.w;
  float s = x0 + x1 + x2 + x3;
#pragma unroll
  for (int o = 16; o > 0; o >>= 1) s += __shfl_xor(s, o, 32);
  float mean = s * (1.0f / 128.0f);
  float d0 = x0 - mean, d1 = x1 - mean, d2 = x2 - mean, d3 = x3 - mean;
  float ss = d0 * d0 + d1 * d1 + d2 * d2 + d3 * d3;
#pragma unroll
  for (int o = 16; o > 0; o >>= 1) ss += __shfl_xor(ss, o, 32);
  float rstd = rsqrtf(ss * (1.0f / 128.0f) + 1e-5f);
  float4 g4 = ((const float4*)g)[lane];
  float4 t4 = ((const float4*)bt)[lane];
  float y0 = d0 * rstd * g4.x + t4.x;
  float y1 = d1 * rstd * g4.y + t4.y;
  float y2 = d2 * rstd * g4.z + t4.z;
  float y3 = d3 * rstd * g4.w + t4.w;
  ((float4*)(yf + (size_t)row * C_))[lane] = make_float4(y0, y1, y2, y3);
  if (yh) {
    half_t* hp = yh + (size_t)row * C_ + lane * 4;
    hp[0] = (half_t)y0; hp[1] = (half_t)y1; hp[2] = (half_t)y2; hp[3] = (half_t)y3;
  }
}

// ---------------- FFN GEMM1 + ReLU: [32400,128]x[128,1024] ---------------
// grid (2025, 2), block 256 (8 waves x 64 cols)
__global__ void k_ffn1(const half_t* srcln16, const half_t* pack, const float* b1,
                       half_t* hid16) {
  int lane = threadIdx.x & 31, wid = threadIdx.x >> 5;
  int m = lane & 15, hf = lane >> 4;
  int r0 = blockIdx.x * 16;
  int nb = blockIdx.y * 512 + wid * 64;
  const half_t* P = pack + PK_W1;
  v8f acc[4] = {};
#pragma unroll
  for (int kt = 0; kt < 4; kt++) {
    v16h a = load_a16(srcln16, r0 + m, C_, kt, hf);
#pragma unroll
    for (int j = 0; j < 4; j++)
      acc[j] = wmma_f16(a, load_bpk(P, kt, (nb >> 4) + j, 4, lane), acc[j]);
  }
#pragma unroll
  for (int j = 0; j < 4; j++) {
    int n = nb + j * 16 + (lane & 15);
    float bias = b1[n];
#pragma unroll
    for (int r = 0; r < 8; r++) {
      float v = fmaxf(acc[j][r] + bias, 0.0f);
      hid16[(size_t)(r0 + hf * 8 + r) * DFF_ + n] = (half_t)v;
    }
  }
}

// ---------------- FFN GEMM2: [32400,1024]x[1024,128] ---------------------
// grid 2025, block 64
__global__ void k_ffn2(const half_t* hid16, const half_t* pack, const float* b2,
                       float* ffn) {
  int lane = threadIdx.x & 31, wid = threadIdx.x >> 5;
  int m = lane & 15, hf = lane >> 4;
  int r0 = blockIdx.x * 16, nb = wid * 64;
  const half_t* P = pack + PK_W2;
  v8f acc[4] = {};
#pragma unroll
  for (int kt = 0; kt < 32; kt++) {
    v16h a = load_a16(hid16, r0 + m, DFF_, kt, hf);
#pragma unroll
    for (int j = 0; j < 4; j++)
      acc[j] = wmma_f16(a, load_bpk(P, kt, (nb >> 4) + j, 32, lane), acc[j]);
  }
#pragma unroll
  for (int j = 0; j < 4; j++) {
    int n = nb + j * 16 + (lane & 15);
    float bias = b2[n];
#pragma unroll
    for (int r = 0; r < 8; r++)
      ffn[(size_t)(r0 + hf * 8 + r) * C_ + n] = acc[j][r] + bias;
  }
}

// ---------------- driver ----------------
extern "C" void kernel_launch(void* const* d_in, const int* in_sizes, int n_in,
                              void* d_out, int out_size, void* d_ws, size_t ws_size,
                              hipStream_t stream) {
  const float* src_all = (const float*)d_in[0];
  const float* refp    = (const float*)d_in[1];
  const float* conv_w  = (const float*)d_in[3];
  const float* conv_b  = (const float*)d_in[4];
  const float* W_val   = (const float*)d_in[5];
  const float* b_val   = (const float*)d_in[6];
  const float* W_off   = (const float*)d_in[7];
  const float* b_off   = (const float*)d_in[8];
  const float* W_attn  = (const float*)d_in[9];
  const float* b_attn  = (const float*)d_in[10];
  const float* W_out   = (const float*)d_in[11];
  const float* b_out   = (const float*)d_in[12];
  const float* ln1_g   = (const float*)d_in[13];
  const float* ln1_b   = (const float*)d_in[14];
  const float* W1      = (const float*)d_in[15];
  const float* b1      = (const float*)d_in[16];
  const float* W2      = (const float*)d_in[17];
  const float* b2      = (const float*)d_in[18];
  const float* ln2_g   = (const float*)d_in[19];
  const float* ln2_b   = (const float*)d_in[20];

  char* ws = (char*)d_ws;
  half_t* pack    = (half_t*)(ws + WS_PACK);
  half_t* src16   = (half_t*)(ws + WS_SRC16);
  half_t* q16     = (half_t*)(ws + WS_Q16);
  float*  value   = (float*) (ws + WS_VALUE);
  float*  offatt  = (float*) (ws + WS_OFFATT);
  half_t* out16   = (half_t*)(ws + WS_OUT16);
  float*  src2    = (float*) (ws + WS_SRC2);
  float*  srcf    = (float*) (ws + WS_SRCF);
  half_t* srcln16 = (half_t*)(ws + WS_SRCLN);
  half_t* hid16   = (half_t*)(ws + WS_HID16);
  float*  ffn     = (float*) (ws + WS_FFN);

  k_prep<<<2352, 256, 0, stream>>>(conv_w, W_val, W_off, W_attn, W_out, W1, W2, pack);
  k_cvt<<<12150, 256, 0, stream>>>(src_all, src16, NTOK * C_);
  k_conv<<<dim3(2025, 1, 3), 64, 0, stream>>>(src16, pack, conv_b, q16);
  k_val<<<6075, 64, 0, stream>>>(src16, pack, b_val, value);
  k_offattn<<<6075, 64, 0, stream>>>(q16, pack, b_off, b_attn, offatt);
  k_sample<<<1013, 256, 0, stream>>>(offatt, value, refp, out16);
  k_outproj<<<2025, 64, 0, stream>>>(out16, pack, b_out, src2);
  k_ln<<<4050, 256, 0, stream>>>(src_all, src2, ln1_g, ln1_b, srcf, srcln16);
  k_ffn1<<<dim3(2025, 2), 256, 0, stream>>>(srcln16, pack, b1, hid16);
  k_ffn2<<<2025, 64, 0, stream>>>(hid16, pack, b2, ffn);
  k_ln<<<4050, 256, 0, stream>>>(srcf, ffn, ln2_g, ln2_b, (float*)d_out, (half_t*)nullptr);
}
